// Attention_57449482551798
// MI455X (gfx1250) — compile-verified, matrix operand order checked by reading, not run on hardware
//
#include <hip/hip_runtime.h>

// Attention with 2D RoPE, flash-attention style, CDNA5 (gfx1250).
// Pass 1: RoPE + f16 conversion once per element into workspace (d_ws).
// Pass 2: flash attention with WMMA f16; K/V tiles staged by the Tensor Data
//         Mover (TENSORcnt), Q staged by async load-to-LDS (ASYNCcnt),
//         double-buffered K/V in LDS.
// B=8, H=8, N=1024 (32x32 grid), D=256.

#define GRID_W 32
#define NTOK   1024
#define DIM    256
#define BH     64          // B*H
#define BM     16          // query rows per block
#define BN     32          // key rows per tile
#define NKT    (NTOK / BN) // 32 K-tiles
#define SCALE  0.0625f     // 1/sqrt(256)

typedef _Float16 v16h __attribute__((ext_vector_type(16)));
typedef _Float16 v8h  __attribute__((ext_vector_type(8)));
typedef float    v8f  __attribute__((ext_vector_type(8)));
typedef unsigned u32x4 __attribute__((ext_vector_type(4)));
typedef unsigned u32x8 __attribute__((ext_vector_type(8)));

// ---------------- CDNA5 async load-to-LDS (ASYNCcnt) ----------------
__device__ __forceinline__ void async_cp16(unsigned lds_off, const void* gsrc) {
    asm volatile("global_load_async_to_lds_b128 %0, %1, off"
                 :: "v"(lds_off), "v"(gsrc)
                 : "memory");
}

__device__ __forceinline__ void wait_async0() {
#if __has_builtin(__builtin_amdgcn_s_wait_asynccnt)
    __builtin_amdgcn_s_wait_asynccnt(0);
#else
    asm volatile("s_wait_asynccnt 0" ::: "memory");
#endif
}

__device__ __forceinline__ void wait_tensor0() {
#if __has_builtin(__builtin_amdgcn_s_wait_tensorcnt)
    __builtin_amdgcn_s_wait_tensorcnt(0);
#else
    asm volatile("s_wait_tensorcnt 0" ::: "memory");
#endif
}

// Generic LDS pointer -> hardware LDS byte address (flat addr[31:0]).
__device__ __forceinline__ unsigned lds_addr(const void* p) {
    return (unsigned)(size_t)p;
}

// ---------------- CDNA5 Tensor Data Mover (TENSORcnt) ----------------
// 2D tile load, 16-bit elements. D# groups per ISA 8.3/8.4; 2-group form
// (VADDR2/VADDR3 = NULL) is valid for tensors up to 2D.
__device__ __forceinline__ void tdm_load_2d(unsigned lds, const void* gaddr,
                                            unsigned tile_d0, unsigned tile_d1,
                                            unsigned tensor_d0, unsigned tensor_d1,
                                            unsigned stride0_halves) {
    const size_t ga = (size_t)gaddr;
    u32x4 g0;
    g0[0] = 1u;                                    // count=1, user descriptor
    g0[1] = lds;                                   // lds_addr [63:32]
    g0[2] = (unsigned)ga;                          // global_addr [95:64]
    g0[3] = ((unsigned)(ga >> 32) & 0x01FFFFFFu)   // global_addr [120:96]
            | 0x80000000u;                         // type=2 ("image") [127:126]
    u32x8 g1;
    g1[0] = 1u << 16;                              // data_size=1 -> 2 bytes
    g1[1] = (tensor_d0 & 0xFFFFu) << 16;           // tensor_dim0[15:0] @ 63:48
    g1[2] = (tensor_d0 >> 16) | ((tensor_d1 & 0xFFFFu) << 16);
    g1[3] = (tensor_d1 >> 16) | (tile_d0 << 16);   // tile_dim0 @ 127:112
    g1[4] = tile_d1;                               // tile_dim1; tile_dim2=0
    g1[5] = stride0_halves;                        // tensor_dim0_stride[31:0]
    g1[6] = 0u;                                    // stride0 hi, stride1 lo
    g1[7] = 0u;                                    // stride1 hi
    asm volatile("tensor_load_to_lds %0, %1" :: "s"(g0), "s"(g1) : "memory");
}

// Build a 16-half WMMA fragment from two contiguous 8-half (16B) runs in LDS.
__device__ __forceinline__ v16h frag_from_lds(const _Float16* rowbase, int loK) {
    v8h lo = *(const v8h*)(rowbase + loK);
    v8h hi = *(const v8h*)(rowbase + loK + 16);
    return __builtin_shufflevector(lo, hi, 0,1,2,3,4,5,6,7,8,9,10,11,12,13,14,15);
}

// 2D RoPE angle for token tok (on 32x32 grid), complex-pair index p in [0,128).
__device__ __forceinline__ void rope_cs(int tok, int p, float* c, float* s) {
    int   x    = tok & (GRID_W - 1);
    int   y    = tok >> 5;
    int   j    = p & 63;
    float pos  = (p < 64) ? (float)x : (float)y;
    float freq = __expf((float)j * -0.14391157f);   // -ln(10000)/64
    __sincosf(pos * freq, s, c);
}

__device__ __forceinline__ v8f wmma_f16(v16h a, v16h b, v8f c) {
    return __builtin_amdgcn_wmma_f32_16x16x32_f16(false, a, false, b, (short)0, c,
                                                  false, false);
}

__device__ __forceinline__ unsigned pack_h2(float a, float b) {
    union { _Float16 h[2]; unsigned u; } pk;
    pk.h[0] = (_Float16)a;
    pk.h[1] = (_Float16)b;
    return pk.u;
}

// ================= Pass 1: RoPE + f16 conversion (once) =================
__global__ __launch_bounds__(256, 1)
void prep_kernel(const float* __restrict__ Q,
                 const float* __restrict__ K,
                 const float* __restrict__ V,
                 _Float16* __restrict__ QR,
                 _Float16* __restrict__ KR,
                 _Float16* __restrict__ VT) {
    const int bh      = blockIdx.y;
    const int tokBase = blockIdx.x * 32;
    const int tid     = threadIdx.x;
    const size_t base = ((size_t)bh * NTOK + tokBase) * DIM;
    _Float16* vt = VT + (size_t)bh * DIM * NTOK;

    #pragma unroll
    for (int i = 0; i < 16; ++i) {          // 32 tokens * 128 pairs / 256 thr
        int pidx = tid + 256 * i;
        int row  = pidx >> 7;
        int p    = pidx & 127;
        int tok  = tokBase + row;
        float c, s;
        rope_cs(tok, p, &c, &s);
        size_t off = base + (size_t)row * DIM + 2 * p;

        float2 q2 = *(const float2*)(Q + off);
        *(unsigned*)(QR + off) = pack_h2((q2.x * c - q2.y * s) * SCALE,
                                         (q2.x * s + q2.y * c) * SCALE);
        float2 k2 = *(const float2*)(K + off);
        *(unsigned*)(KR + off) = pack_h2(k2.x * c - k2.y * s,
                                         k2.x * s + k2.y * c);
        float2 v2 = *(const float2*)(V + off);
        vt[(size_t)(2 * p) * NTOK + tok]     = (_Float16)v2.x;
        vt[(size_t)(2 * p + 1) * NTOK + tok] = (_Float16)v2.y;
    }
}

// ================= Pass 2: flash attention (WMMA f16) =================
__global__ __launch_bounds__(256, 1)
void fa_kernel(const _Float16* __restrict__ QR,
               const _Float16* __restrict__ KR,
               const _Float16* __restrict__ VT,
               float* __restrict__ O) {
    __shared__ _Float16 Qs[BM][DIM];          // 8 KB
    __shared__ _Float16 Ks[2][BN][DIM];       // 2 x 16 KB (double buffer)
    __shared__ _Float16 VsT[2][DIM][BN];      // 2 x 16 KB (double buffer)
    __shared__ _Float16 Ps[BM][BN];           // 1 KB

    const int qTile = blockIdx.x;             // 0..63
    const int bh    = blockIdx.y;             // 0..63
    const int tid   = threadIdx.x;
    const int lane  = tid & 31;
    const int wave  = tid >> 5;
    const int qBase = qTile * BM;

    const _Float16* Qp  = QR + ((size_t)bh * NTOK + qBase) * DIM;
    const _Float16* Kp  = KR + (size_t)bh * NTOK * DIM;
    const _Float16* VTp = VT + (size_t)bh * DIM * NTOK;
    float*          Op  = O  + ((size_t)bh * NTOK + qBase) * DIM;

    // ---- async-stage Q tile (8 KB contiguous = 512 x 16B chunks) ----
    {
        const unsigned qdst = lds_addr(&Qs[0][0]);
        #pragma unroll
        for (int i = 0; i < 2; ++i) {
            int c = tid + 256 * i;
            async_cp16(qdst + c * 16, Qp + c * 8);
        }
    }

    // ---- one TDM issue pair per K/V tile (wave 0 only; EXEC ignored) ----
    auto issue_tile = [&](int kt, int buf) {
        // K tile: 16 KB contiguous -> 1 x 8192-half tile
        tdm_load_2d(lds_addr(&Ks[buf][0][0]), Kp + (size_t)kt * BN * DIM,
                    /*tile_d0=*/BN * DIM, /*tile_d1=*/1,
                    /*tensor_d0=*/BN * DIM, /*tensor_d1=*/1,
                    /*stride0=*/BN * DIM);
        // V^T tile: 256 rows x 32 halves, row stride 1024 halves
        tdm_load_2d(lds_addr(&VsT[buf][0][0]), VTp + (size_t)kt * BN,
                    /*tile_d0=*/BN, /*tile_d1=*/DIM,
                    /*tensor_d0=*/NTOK, /*tensor_d1=*/DIM,
                    /*stride0=*/NTOK);
    };
    if (wave == 0) issue_tile(0, 0);
    wait_async0();                            // this wave's Q chunks landed

    // Fragment addressing (ISA 7.12.2, 16-bit): lanes 0-15 hold runs
    // {K 0-7, K 16-23}, lanes 16-31 hold {K 8-15, K 24-31}.
    const int fRow = lane & 15;
    const int loK  = (lane < 16) ? 0 : 8;

    v16h qf[8];

    // Online-softmax state (replicated per wave; lanes 0-15 <-> rows 0-7,
    // lanes 16-31 <-> rows 8-15, matching the WMMA C layout).
    float mrow[8], lrow[8];
    #pragma unroll
    for (int r = 0; r < 8; ++r) { mrow[r] = -1e30f; lrow[r] = 0.0f; }

    // Each wave owns 32 columns of D for the PV product.
    const int c0 = wave * 32;
    const int c1 = c0 + 16;
    v8f acc0 = {}, acc1 = {};

    for (int kt = 0; kt < NKT; ++kt) {
        if (wave == 0) wait_tensor0();        // tile kt landed in LDS
        __syncthreads();                      // publish to all waves
        if (kt + 1 < NKT && wave == 0)        // overlap DMA with compute
            issue_tile(kt + 1, (kt + 1) & 1);
        const int buf = kt & 1;

        if (kt == 0) {          // Q A-fragments live in registers (8 x 8 VGPRs)
            #pragma unroll
            for (int kc = 0; kc < 8; ++kc)
                qf[kc] = frag_from_lds(&Qs[fRow][kc * 32], loK);
        }

        // ---- S = (Q/16) . K^T : two 16x16 tiles over k=256 ----
        v8f Sa = {}, Sb = {};
        #pragma unroll
        for (int kc = 0; kc < 8; ++kc) {
            v16h kb0 = frag_from_lds(&Ks[buf][fRow][kc * 32], loK);
            v16h kb1 = frag_from_lds(&Ks[buf][fRow + 16][kc * 32], loK);
            Sa = wmma_f16(qf[kc], kb0, Sa);
            Sb = wmma_f16(qf[kc], kb1, Sb);
        }

        // ---- online softmax (half-wave shuffle reductions per row) ----
        float alpha[8];
        #pragma unroll
        for (int r = 0; r < 8; ++r) {
            float mx = fmaxf(Sa[r], Sb[r]);
            #pragma unroll
            for (int d = 8; d >= 1; d >>= 1)
                mx = fmaxf(mx, __shfl_xor(mx, d, 16));
            float mnew = fmaxf(mrow[r], mx);
            alpha[r]   = __expf(mrow[r] - mnew);
            mrow[r]    = mnew;
            float pa = __expf(Sa[r] - mnew);
            float pb = __expf(Sb[r] - mnew);
            Sa[r] = pa; Sb[r] = pb;
            float rs = pa + pb;
            #pragma unroll
            for (int d = 8; d >= 1; d >>= 1)
                rs += __shfl_xor(rs, d, 16);
            lrow[r] = lrow[r] * alpha[r] + rs;
        }

        // S/softmax identical in every wave: wave 0 publishes P as f16.
        if (wave == 0) {
            const int rbase = (lane >> 4) << 3;   // 0 or 8
            #pragma unroll
            for (int r = 0; r < 8; ++r) {
                Ps[rbase + r][lane & 15]        = (_Float16)Sa[r];
                Ps[rbase + r][(lane & 15) + 16] = (_Float16)Sb[r];
            }
        }
        __syncthreads();

        // ---- rescale accumulators, then O += P . V for this wave's cols ----
        #pragma unroll
        for (int r = 0; r < 8; ++r) { acc0[r] *= alpha[r]; acc1[r] *= alpha[r]; }

        v16h pf  = frag_from_lds(&Ps[fRow][0], loK);
        v16h vb0 = frag_from_lds(&VsT[buf][c0 + fRow][0], loK);
        v16h vb1 = frag_from_lds(&VsT[buf][c1 + fRow][0], loK);
        acc0 = wmma_f16(pf, vb0, acc0);
        acc1 = wmma_f16(pf, vb1, acc1);
    }

    // ---- normalize and store ----
    const int rbase = (lane >> 4) << 3;           // 0 or 8
    #pragma unroll
    for (int r = 0; r < 8; ++r) {
        float inv = 1.0f / lrow[r];
        Op[(size_t)(rbase + r) * DIM + c0 + (lane & 15)] = acc0[r] * inv;
        Op[(size_t)(rbase + r) * DIM + c1 + (lane & 15)] = acc1[r] * inv;
    }
}

extern "C" void kernel_launch(void* const* d_in, const int* in_sizes, int n_in,
                              void* d_out, int out_size, void* d_ws, size_t ws_size,
                              hipStream_t stream) {
    const float* Q = (const float*)d_in[0];
    const float* K = (const float*)d_in[1];
    const float* V = (const float*)d_in[2];
    float*       O = (float*)d_out;

    const size_t perTensor = (size_t)BH * NTOK * DIM;   // 16,777,216 elements
    _Float16* wsQ = (_Float16*)d_ws;
    _Float16* wsK = wsQ + perTensor;
    _Float16* wsV = wsK + perTensor;

    dim3 pgrid(NTOK / 32, BH);    // 32 x 64 blocks
    prep_kernel<<<pgrid, 256, 0, stream>>>(Q, K, V, wsQ, wsK, wsV);

    dim3 agrid(NTOK / BM, BH);    // 64 q-tiles x 64 (b,h)
    fa_kernel<<<agrid, 256, 0, stream>>>(wsQ, wsK, wsV, O);
}
